// SkeletonGCNAutoencoder_28595892256779
// MI455X (gfx1250) — matrix operand order repaired and checked by reading, not compiled, not run on hardware
//
#include <hip/hip_runtime.h>
#include <hip/hip_bf16.h>

typedef __attribute__((ext_vector_type(16))) _Float16 v16h;
typedef __attribute__((ext_vector_type(8)))  float    v8f;

#define J    25
#define JP   32
#define CIN  3
#define HID  64
#define LAT  32

// ---- workspace layout ----
#define WS_B23_F  0            // 64 f32
#define WS_H_BYTE 256          // f16 region start (16B aligned)
// f16 offsets (halves):
#define WS_M23_H  0            // 1024: M23 padded [32][32] row-major
#define WS_W23_H  1024         // 4096: W23^T layout [e][d] (= col-major W23), stride 64
#define WS_TAB_H  5120         // tables copied to LDS:
#define T_M1  0                //   M1h  [32][32] row-major (1024)
#define T_M4  1024             //   M4h  [32][32] row-major (1024)
#define T_W1  2048             //   W1t  [d=64][k=32], k<3 -> W1[k][d], k==3 -> b1[d] (2048)
#define T_W4  4096             //   W4t  [n=16][k=64], n<3 -> W4[k][n] (1024)
#define T_TOT 5120

union V16HU { v16h v; uint4 q[2]; };

__device__ inline v16h ld_frag(const _Float16* p0, const _Float16* p1) {
  V16HU u;
  u.q[0] = *(const uint4*)p0;
  u.q[1] = *(const uint4*)p1;
  return u.v;
}

__device__ inline void st_pair(_Float16* p, float a, float b) {
  auto v = __builtin_amdgcn_cvt_pkrtz(a, b);   // v_cvt_pk_rtz_f16_f32
  __builtin_memcpy(p, &v, 4);
}

__device__ inline float relu1(float x) {       // v_med3_f32 x, 0, +inf (1 op)
  return __builtin_amdgcn_fmed3f(x, 0.f, __builtin_inff());
}

__device__ inline void wave_lds_fence() {
  __asm__ volatile("s_wait_dscnt 0x0" ::: "memory");
}

__device__ inline void softmax_row(const float* __restrict__ A,
                                   float* __restrict__ dst, int r) {
  float m = -1e30f;
  for (int j = 0; j < J; ++j) m = fmaxf(m, A[r * J + j]);
  float e[J]; float s = 0.f;
  for (int j = 0; j < J; ++j) { e[j] = __expf(A[r * J + j] - m); s += e[j]; }
  float inv = 1.f / s;
  for (int j = 0; j < J; ++j) dst[r * J + j] = e[j] * inv;
}

// ---------- setup: softmaxes + fused/padded f16 operators ----------
__global__ void gcn_ae_setup(const float* __restrict__ A1, const float* __restrict__ A2,
                             const float* __restrict__ A3, const float* __restrict__ A4,
                             const float* __restrict__ W1, const float* __restrict__ b1,
                             const float* __restrict__ W2, const float* __restrict__ b2,
                             const float* __restrict__ W3, const float* __restrict__ b3,
                             const float* __restrict__ W4,
                             float* __restrict__ wsf, _Float16* __restrict__ wsh) {
  __shared__ float sM[4][J * J];
  const int tid = threadIdx.x;

  if (tid < J)                       softmax_row(A1, sM[0], tid);
  else if (tid >= 32 && tid < 32+J)  softmax_row(A2, sM[1], tid - 32);
  else if (tid >= 64 && tid < 64+J)  softmax_row(A3, sM[2], tid - 64);
  else if (tid >= 96 && tid < 96+J)  softmax_row(A4, sM[3], tid - 96);
  __syncthreads();

  // M23 = softmax(A3) @ softmax(A2), padded [32][32] f16 row-major
  for (int idx = tid; idx < JP * JP; idx += 256) {
    int i = idx >> 5, j = idx & 31;
    float v = 0.f;
    if (i < J && j < J)
      for (int k = 0; k < J; ++k) v += sM[2][i * J + k] * sM[1][k * J + j];
    wsh[WS_M23_H + idx] = (_Float16)v;
  }
  // W23^T layout: wsh[e*64+d] = sum_k W2[d][k] W3[k][e]
  for (int idx = tid; idx < HID * HID; idx += 256) {
    int e = idx >> 6, d = idx & 63;
    float v = 0.f;
    for (int k = 0; k < LAT; ++k) v += W2[d * LAT + k] * W3[k * HID + e];
    wsh[WS_W23_H + idx] = (_Float16)v;
  }
  // M1h / M4h padded [32][32] row-major
  for (int idx = tid; idx < JP * JP; idx += 256) {
    int i = idx >> 5, j = idx & 31;
    float v1 = 0.f, v4 = 0.f;
    if (i < J && j < J) { v1 = sM[0][i * J + j]; v4 = sM[3][i * J + j]; }
    wsh[WS_TAB_H + T_M1 + idx] = (_Float16)v1;
    wsh[WS_TAB_H + T_M4 + idx] = (_Float16)v4;
  }
  // W1t[d*32+k]: k<3 -> W1[k][d]; k==3 -> b1[d] (homogeneous bias); else 0
  for (int idx = tid; idx < HID * JP; idx += 256) {
    int d = idx >> 5, k = idx & 31;
    float v = 0.f;
    if (k < CIN)       v = W1[k * HID + d];
    else if (k == CIN) v = b1[d];
    wsh[WS_TAB_H + T_W1 + idx] = (_Float16)v;
  }
  // W4t[n*64+k] = (n<3) ? W4[k][n] : 0
  for (int idx = tid; idx < 16 * HID; idx += 256) {
    int n = idx >> 6, k = idx & 63;
    float v = (n < CIN) ? W4[k * CIN + n] : 0.f;
    wsh[WS_TAB_H + T_W4 + idx] = (_Float16)v;
  }
  // b23[e] = b3[e] + sum_k b2[k] W3[k][e]
  if (tid < HID) {
    float v = b3[tid];
    for (int k = 0; k < LAT; ++k) v += b2[k] * W3[k * HID + tid];
    wsf[WS_B23_F + tid] = v;
  }
}

// ---------- fused autoencoder: every stage on WMMA, contiguous epilogues ----
__global__ __launch_bounds__(256)
void gcn_ae_fused(const float* __restrict__ x, const float* __restrict__ b4,
                  const float* __restrict__ wsf, const _Float16* __restrict__ wsh,
                  float* __restrict__ out, int NT, int nWavesTotal) {
  __shared__ __align__(16) _Float16 tabs[T_TOT];
  __shared__ float cb23[HID];
  __shared__ float cb4[4];
  // per-wave halves: xcm[16][32] | ut[32][32] | big[2048] (hcm->ttm->hd) | pcm[16][32]
  __shared__ __align__(16) _Float16 wv[8][4096];
#define O_XCM 0
#define O_UT  512
#define O_BIG 1536
#define O_PCM 3584

  const int tid = threadIdx.x;
  for (int i = tid; i < T_TOT; i += 256) tabs[i] = wsh[WS_TAB_H + i];
  for (int i = tid; i < HID; i += 256) cb23[i] = wsf[WS_B23_F + i];
  if (tid < 3) cb4[tid] = b4[tid];
  __syncthreads();

  const int lane = tid & 31;
  const int wid  = tid >> 5;
  const int hi   = lane >> 4;
  const int l15  = lane & 15;
  const int rb   = 8 * hi;             // C-layout row base inside a 16-tile
  const int ka   = hi ? 8 : 0;         // A-fragment k base
  const int kb   = hi ? 8 : 0;         // B-fragment k base
  _Float16* xcm = &wv[wid][O_XCM];     // x^T [c][j]
  _Float16* ut  = &wv[wid][O_UT];      // u   [i][c]
  _Float16* big = &wv[wid][O_BIG];     // hcm[d][j] -> ttm[i][d] -> hd[i][e]
  _Float16* pcm = &wv[wid][O_PCM];     // p^T [e'][i]

  // one-time pads (never rewritten in loop)
  for (int c = CIN + 1; c < 16; ++c) xcm[c * JP + lane] = (_Float16)0.f;
  xcm[CIN * JP + lane] = (_Float16)1.f;              // homogeneous bias row
  for (int c = 16; c < 32; ++c) ut[lane * JP + c] = (_Float16)0.f;

  // persistent middle-operator fragments
  v16h m23tb[2];                                     // B of M23^T (= M23 rows)
#pragma unroll
  for (int n = 0; n < 2; ++n) {
    const _Float16* p = wsh + WS_M23_H + (16 * n + l15) * JP + kb;
    m23tb[n] = ld_frag(p, p + 16);
  }
  v16h aw[4][2];                                     // A of W23^T
#pragma unroll
  for (int m = 0; m < 4; ++m)
#pragma unroll
    for (int ks = 0; ks < 2; ++ks) {
      const _Float16* p = wsh + WS_W23_H + (l15 + 16 * m) * HID + ks * 32 + ka;
      aw[m][ks] = ld_frag(p, p + 16);
    }

  const int gw = blockIdx.x * 8 + wid;
  for (int nt = gw; nt < NT; nt += nWavesTotal) {
    // ===== stage 1: u^T = x^T @ M1^T ; h = relu(u @ W1t) (2 + 8 WMMA) =====
    {
      const float* xr = x + (size_t)nt * (J * CIN);
      float x0 = 0.f, x1 = 0.f, x2 = 0.f;
      if (lane < J) { x0 = xr[lane*3+0]; x1 = xr[lane*3+1]; x2 = xr[lane*3+2]; }
      xcm[0 * JP + lane] = (_Float16)x0;
      xcm[1 * JP + lane] = (_Float16)x1;
      xcm[2 * JP + lane] = (_Float16)x2;
    }
    wave_lds_fence();
    v16h xa;
    { const _Float16* p = xcm + l15 * JP + ka; xa = ld_frag(p, p + 16); }
    v16h m1b[2];
#pragma unroll
    for (int n = 0; n < 2; ++n) {
      const _Float16* p = tabs + T_M1 + (16 * n + l15) * JP + kb;
      m1b[n] = ld_frag(p, p + 16);
    }
#pragma unroll
    for (int n = 0; n < 2; ++n) {                    // u^T tiles -> ut[i][c]
      v8f z = {0.f,0.f,0.f,0.f,0.f,0.f,0.f,0.f};
      v8f u = __builtin_amdgcn_wmma_f32_16x16x32_f16(
          false, xa, false, m1b[n], (short)0, z, false, false);
      _Float16* bp_ = ut + (16 * n + l15) * JP + rb;
      st_pair(bp_ + 0, u[0], u[1]); st_pair(bp_ + 2, u[2], u[3]);
      st_pair(bp_ + 4, u[4], u[5]); st_pair(bp_ + 6, u[6], u[7]);
    }
    wave_lds_fence();
    v16h ua[2], w1b[4];
#pragma unroll
    for (int m = 0; m < 2; ++m) {
      const _Float16* p = ut + (l15 + 16 * m) * JP + ka;
      ua[m] = ld_frag(p, p + 16);
    }
#pragma unroll
    for (int n = 0; n < 4; ++n) {
      const _Float16* p = tabs + T_W1 + (16 * n + l15) * JP + kb;
      w1b[n] = ld_frag(p, p + 16);
    }
    wave_lds_fence();
#pragma unroll
    for (int m = 0; m < 2; ++m)                      // h tiles -> hcm[d][j]
#pragma unroll
      for (int n = 0; n < 4; ++n) {
        v8f z = {0.f,0.f,0.f,0.f,0.f,0.f,0.f,0.f};
        v8f h = __builtin_amdgcn_wmma_f32_16x16x32_f16(
            false, ua[m], false, w1b[n], (short)0, z, false, false);
        _Float16* bp_ = big + (16 * n + l15) * JP + 16 * m + rb;
        st_pair(bp_ + 0, relu1(h[0]), relu1(h[1]));
        st_pair(bp_ + 2, relu1(h[2]), relu1(h[3]));
        st_pair(bp_ + 4, relu1(h[4]), relu1(h[5]));
        st_pair(bp_ + 6, relu1(h[6]), relu1(h[7]));
      }
    wave_lds_fence();

    // ===== stage 2 mix: T^T = h^T @ M23^T (8 WMMA) =====
    v16h ha[4];
#pragma unroll
    for (int mt = 0; mt < 4; ++mt) {
      const _Float16* p = big + (l15 + 16 * mt) * JP + ka;
      ha[mt] = ld_frag(p, p + 16);
    }
    wave_lds_fence();                                // all reads before alias stores
#pragma unroll
    for (int mt = 0; mt < 4; ++mt)
#pragma unroll
      for (int n = 0; n < 2; ++n) {
        v8f z = {0.f,0.f,0.f,0.f,0.f,0.f,0.f,0.f};
        v8f t = __builtin_amdgcn_wmma_f32_16x16x32_f16(
            false, ha[mt], false, m23tb[n], (short)0, z, false, false);
        _Float16* bp_ = big + (16 * n + l15) * HID + 16 * mt + rb;   // ttm[i][d]
        st_pair(bp_ + 0, t[0], t[1]); st_pair(bp_ + 2, t[2], t[3]);
        st_pair(bp_ + 4, t[4], t[5]); st_pair(bp_ + 6, t[6], t[7]);
      }
    wave_lds_fence();

    // ===== stage 2 proj: Hd^T = W23^T @ T^T + b23, relu (16 WMMA) =====
    v16h tb[2][2];
#pragma unroll
    for (int n = 0; n < 2; ++n)
#pragma unroll
      for (int ks = 0; ks < 2; ++ks) {
        const _Float16* p = big + (16 * n + l15) * HID + ks * 32 + kb;
        tb[n][ks] = ld_frag(p, p + 16);
      }
    wave_lds_fence();                                // all reads before alias stores
#pragma unroll
    for (int m = 0; m < 4; ++m) {
      float bb[8];
#pragma unroll
      for (int r = 0; r < 8; ++r) bb[r] = cb23[16 * m + rb + r];
#pragma unroll
      for (int n = 0; n < 2; ++n) {
        v8f z = {0.f,0.f,0.f,0.f,0.f,0.f,0.f,0.f};
        v8f a0 = __builtin_amdgcn_wmma_f32_16x16x32_f16(
            false, aw[m][0], false, tb[n][0], (short)0, z, false, false);
        a0 = __builtin_amdgcn_wmma_f32_16x16x32_f16(
            false, aw[m][1], false, tb[n][1], (short)0, a0, false, false);
        _Float16* bp_ = big + (16 * n + l15) * HID + 16 * m + rb;    // hd[i][e]
        st_pair(bp_ + 0, relu1(a0[0]+bb[0]), relu1(a0[1]+bb[1]));
        st_pair(bp_ + 2, relu1(a0[2]+bb[2]), relu1(a0[3]+bb[3]));
        st_pair(bp_ + 4, relu1(a0[4]+bb[4]), relu1(a0[5]+bb[5]));
        st_pair(bp_ + 6, relu1(a0[6]+bb[6]), relu1(a0[7]+bb[7]));
      }
    }
    wave_lds_fence();

    // ===== stage 3: p = h_dec @ W4t ; out = M4 @ p + b4 (4 + 2 WMMA) =====
    v16h at3[2][2], w4b[2], m4a[2];
#pragma unroll
    for (int m = 0; m < 2; ++m) {
#pragma unroll
      for (int ks = 0; ks < 2; ++ks) {
        const _Float16* p = big + (l15 + 16 * m) * HID + ks * 32 + ka;
        at3[m][ks] = ld_frag(p, p + 16);
      }
      const _Float16* pm = tabs + T_M4 + (l15 + 16 * m) * JP + ka;
      m4a[m] = ld_frag(pm, pm + 16);
    }
#pragma unroll
    for (int ks = 0; ks < 2; ++ks) {
      const _Float16* p = tabs + T_W4 + l15 * HID + ks * 32 + kb;
      w4b[ks] = ld_frag(p, p + 16);
    }
#pragma unroll
    for (int m = 0; m < 2; ++m) {
      v8f z = {0.f,0.f,0.f,0.f,0.f,0.f,0.f,0.f};
      v8f p0 = __builtin_amdgcn_wmma_f32_16x16x32_f16(
          false, at3[m][0], false, w4b[0], (short)0, z, false, false);
      p0 = __builtin_amdgcn_wmma_f32_16x16x32_f16(
          false, at3[m][1], false, w4b[1], (short)0, p0, false, false);
      _Float16* bp_ = pcm + l15 * JP + 16 * m + rb;                  // p^T[e'][i]
      st_pair(bp_ + 0, p0[0], p0[1]); st_pair(bp_ + 2, p0[2], p0[3]);
      st_pair(bp_ + 4, p0[4], p0[5]); st_pair(bp_ + 6, p0[6], p0[7]);
    }
    wave_lds_fence();
    v16h bp;
    { const _Float16* p = pcm + l15 * JP + kb; bp = ld_frag(p, p + 16); }
#pragma unroll
    for (int m = 0; m < 2; ++m) {
      v8f z = {0.f,0.f,0.f,0.f,0.f,0.f,0.f,0.f};
      v8f o = __builtin_amdgcn_wmma_f32_16x16x32_f16(
          false, m4a[m], false, bp, (short)0, z, false, false);
      if (l15 < CIN) {
        float* orow = out + (size_t)nt * (J * CIN);
#pragma unroll
        for (int r = 0; r < 8; ++r) {
          const int row = 16 * m + rb + r;
          if (row < J) orow[row * CIN + l15] = o[r] + cb4[l15];
        }
      }
    }
    if (nt + nWavesTotal < NT)
      __builtin_prefetch(x + (size_t)(nt + nWavesTotal) * (J * CIN), 0, 0);
  }
}

extern "C" void kernel_launch(void* const* d_in, const int* in_sizes, int n_in,
                              void* d_out, int out_size, void* d_ws, size_t ws_size,
                              hipStream_t stream) {
  const float* x  = (const float*)d_in[0];
  const float* A1 = (const float*)d_in[1];
  const float* W1 = (const float*)d_in[2];
  const float* b1 = (const float*)d_in[3];
  const float* A2 = (const float*)d_in[4];
  const float* W2 = (const float*)d_in[5];
  const float* b2 = (const float*)d_in[6];
  const float* A3 = (const float*)d_in[7];
  const float* W3 = (const float*)d_in[8];
  const float* b3 = (const float*)d_in[9];
  const float* A4 = (const float*)d_in[10];
  const float* W4 = (const float*)d_in[11];
  const float* b4 = (const float*)d_in[12];
  float*     out = (float*)d_out;
  float*     wsf = (float*)d_ws;
  _Float16*  wsh = (_Float16*)((char*)d_ws + WS_H_BYTE);

  const int NT = in_sizes[0] / (J * CIN);   // 65536

  gcn_ae_setup<<<1, 256, 0, stream>>>(A1, A2, A3, A4, W1, b1, W2, b2, W3, b3, W4,
                                      wsf, wsh);

  const int blocks = 512;                   // 8 waves/block -> 4096 waves
  const int nWaves = blocks * 8;
  gcn_ae_fused<<<blocks, 256, 0, stream>>>(x, b4, wsf, wsh, out, NT, nWaves);
}
